// ReadformerBlock_8710193676346
// MI455X (gfx1250) — matrix operand order repaired and checked by reading, not compiled, not run on hardware
//
#include <hip/hip_runtime.h>
#include <hip/hip_bf16.h>

// ---------------------------------------------------------------------------
// ReadformerBlock for MI455X (gfx1250, wave32, WMMA).
//   1) z_conv = sum_k shift_k(emb) @ M_k + conv_b, * mask     [WMMA bf16]
//   2) h1 = rope(emb) @ W1 + b1                                [WMMA bf16]
//   3) LayerNorm + exact GELU
//   4) h2 = gelu @ W2 + b2, * ff_scale                         [WMMA bf16]
//   5) per-(b,s,order) L1 normalize over D
//   6) 2x FFT long conv (8192-pt, LDS-resident, twiddle table)
//   7) out = v^T @ W_out + b_out, * mask                       [WMMA bf16]
// GEMM data movement:
//   - weight B-tiles: TDM tensor_load_to_lds (one D# per tile, LDS padding
//     reproduces the 80B row stride), waited with s_wait_tensorcnt
//   - bf16 activation A-tiles: GLOBAL_LOAD_ASYNC_TO_LDS_B128 (ASYNCcnt)
//   - fp32 A-tiles: b128 loads + v_cvt_pk_bf16_f32 + ds_store_b128
// Double-buffered LDS; fragments read as 2x ds_load_b128 per lane.
// ---------------------------------------------------------------------------

typedef __attribute__((ext_vector_type(16))) __bf16 v16bf;
typedef __attribute__((ext_vector_type(8)))  __bf16 v8bf;
typedef __attribute__((ext_vector_type(8)))  float  v8f;
typedef __attribute__((ext_vector_type(4)))  unsigned int u32x4;
typedef __attribute__((ext_vector_type(8)))  int i32x8;
typedef __attribute__((ext_vector_type(4)))  int i32x4;

#define BDIM 8
#define SDIM 4096
#define DDIM 256
#define GD   768
#define ND2  512

#define BM 128
#define BN 64
#define BK 32
#define LDA 40   // BK + 8 pad: 80B rows -> conflict-free b128 frag reads
#define LDB 40
#define THREADS 256

// ---- CDNA5 async global->LDS copy (no VGPR round trip) ---------------------
typedef __attribute__((address_space(3))) char lds_char;

__device__ inline unsigned lds_off(const void* p) {
  return (unsigned)(unsigned long long)(lds_char*)p;   // addrspace(3) byte offset
}

__device__ inline void async_ld16(void* dst_lds, const void* src_global) {
  unsigned dso = lds_off(dst_lds);
  asm volatile("global_load_async_to_lds_b128 %0, %1, off"
               :: "v"(dso), "v"(src_global)
               : "memory");
}

__device__ inline void async_wait_all() {
  asm volatile("s_wait_asynccnt 0" ::: "memory");
}

// ---- CDNA5 TDM: one-descriptor DMA of a [tile_n x tile_k] bf16 tile --------
// Loads rows of tile_k bf16 (64B when tile_k=32) with a 16B LDS pad after
// each row (pad_interval=3 -> every 16 DWORDs, pad_amount=3 -> 4 DWORDs),
// matching the LDB=40 row stride. Issued by one wave; TENSORcnt tracked.
__device__ inline void tdm_load_tile_bf16(void* lds_dst, const void* gsrc,
                                          int tile_k, int tile_n,
                                          int kstride_elems, int n_total) {
  unsigned lds = lds_off(lds_dst);
  unsigned long long ga = (unsigned long long)gsrc;
  u32x4 g0;
  g0[0] = 1u;                                            // count=1, user D#
  g0[1] = lds;                                           // lds_addr (bytes)
  g0[2] = (unsigned)(ga & 0xffffffffu);                  // global_addr[31:0]
  g0[3] = (unsigned)((ga >> 32) & 0x01ffffffu) | (2u << 30);  // [56:32] | type=2
  unsigned td0 = (unsigned)kstride_elems;                // tensor_dim0 (elems)
  unsigned td1 = (unsigned)n_total;                      // tensor_dim1 (rows)
  unsigned st0 = (unsigned)kstride_elems;                // dim0 stride (elems)
  i32x8 g1;
  g1[0] = (int)((1u << 16) | (1u << 20) | (3u << 22) | (3u << 25)); // 2B, pad on
  g1[1] = (int)((td0 & 0xffffu) << 16);                  // barrier_addr=0 | td0 lo
  g1[2] = (int)((td0 >> 16) | ((td1 & 0xffffu) << 16));  // td0 hi | td1 lo
  g1[3] = (int)((td1 >> 16) | ((unsigned)tile_k << 16)); // td1 hi | tile_dim0
  g1[4] = (int)(unsigned)tile_n;                         // tile_dim1 | tile_dim2=0
  g1[5] = (int)st0;                                      // dim0 stride [31:0]
  g1[6] = 0;                                             // stride hi | dim1 stride=0
  g1[7] = 0;
  i32x4 z4 = {0, 0, 0, 0};                               // 2D tensor: groups 2/3 unused
  i32x8 z8 = {0, 0, 0, 0, 0, 0, 0, 0};
  __builtin_amdgcn_tensor_load_to_lds(g0, g1, z4, z4, z8, 0);  // 6-arg toolchain
}

__device__ inline void tensor_wait_all() {
  __builtin_amdgcn_s_wait_tensorcnt((short)0);
}

// ---- WMMA fragment loaders (k = 16*(lane/16) + e, same order for A and B) ------------------
__device__ inline v16bf fragA(const __bf16* As, int mbase, int lane) {
  int m  = mbase + (lane & 15);
  int kh = (lane >> 4) << 4;                 // k = kh + e
  const __bf16* p = As + m * LDA + kh;
  v8bf lo = *(const v8bf*)(p);
  v8bf hi = *(const v8bf*)(p + 8);
  return __builtin_shufflevector(lo, hi, 0,1,2,3,4,5,6,7,8,9,10,11,12,13,14,15);
}

__device__ inline v16bf fragB(const __bf16* Bs, int nbase, int lane) {
  int n  = nbase + (lane & 15);
  int kh = (lane >> 4) << 4;                 // k = kh + e
  const __bf16* p = Bs + n * LDB + kh;
  v8bf lo = *(const v8bf*)(p);
  v8bf hi = *(const v8bf*)(p + 8);
  return __builtin_shufflevector(lo, hi, 0,1,2,3,4,5,6,7,8,9,10,11,12,13,14,15);
}

// ---- Generic 128x64 GEMM core: 8 waves, double-buffered, fp32 accumulate ---
template <typename SA, typename SB, typename EP>
__device__ inline void gemm_core(int Ktot, SA stageA, SB stageB, EP epi) {
  __shared__ __attribute__((aligned(16))) __bf16 As[2][BM * LDA];
  __shared__ __attribute__((aligned(16))) __bf16 Bs[2][BN * LDB];
  const int tid = threadIdx.x, lane = tid & 31, wave = tid >> 5;
  const int wm = wave & 3, wn = wave >> 2;

  v8f acc[2][2];
#pragma unroll
  for (int mi = 0; mi < 2; ++mi)
#pragma unroll
    for (int ni = 0; ni < 2; ++ni)
#pragma unroll
      for (int e = 0; e < 8; ++e) acc[mi][ni][e] = 0.f;

  stageA(As[0], 0);
  stageB(Bs[0], 0);
  async_wait_all();
  tensor_wait_all();
  __syncthreads();

  const int nk = Ktot / BK;
  for (int it = 0; it < nk; ++it) {
    const int cur = it & 1;
    if (it + 1 < nk) {                       // prefetch next tile into other buffer
      stageA(As[cur ^ 1], (it + 1) * BK);
      stageB(Bs[cur ^ 1], (it + 1) * BK);
    }

    v16bf af[2], bfm[2];
#pragma unroll
    for (int mi = 0; mi < 2; ++mi) af[mi]  = fragA(As[cur], wm * 32 + mi * 16, lane);
#pragma unroll
    for (int ni = 0; ni < 2; ++ni) bfm[ni] = fragB(Bs[cur], wn * 32 + ni * 16, lane);
#pragma unroll
    for (int mi = 0; mi < 2; ++mi)
#pragma unroll
      for (int ni = 0; ni < 2; ++ni)
        acc[mi][ni] = __builtin_amdgcn_wmma_f32_16x16x32_bf16(
            false, af[mi], false, bfm[ni], (short)0, acc[mi][ni], false, false);

    async_wait_all();                        // async + TDM stores to cur^1 visible
    tensor_wait_all();
    __syncthreads();
  }

#pragma unroll
  for (int mi = 0; mi < 2; ++mi)
#pragma unroll
    for (int ni = 0; ni < 2; ++ni)
#pragma unroll
      for (int r = 0; r < 8; ++r) {
        int ml = wm * 32 + mi * 16 + r + ((lane >> 4) << 3);
        int nl = wn * 32 + ni * 16 + (lane & 15);
        epi(ml, nl, acc[mi][ni][r]);
      }
}

// Stager for transposed bf16 weights WT[N][Ktot]: one TDM descriptor per tile.
__device__ inline void stage_wt(__bf16* Bs, const __bf16* WT, int Ktot, int Ntot,
                                int n0, int k0) {
  if (threadIdx.x < 32)                      // one wave issues the DMA
    tdm_load_tile_bf16(Bs, WT + (size_t)n0 * Ktot + k0, BK, BN, Ktot, Ntot);
}

// Stager for contiguous bf16 activations src[row][Kfull]: 2 async copies/thread.
__device__ inline void stage_act(__bf16* As, const __bf16* act, int Kfull,
                                 int row0, int k0) {
#pragma unroll
  for (int i = 0; i < 2; ++i) {
    int cidx = (int)threadIdx.x + i * THREADS;   // 0..511: 128 rows x 4 chunks
    int r = cidx >> 2, q = cidx & 3;
    const __bf16* src = act + (size_t)(row0 + r) * Kfull + k0 + q * 8;
    async_ld16(As + r * LDA + q * 8, src);
    if (k0 + BK < Kfull) __builtin_prefetch(src + BK, 0, 3);  // global_prefetch_b8
  }
}

__device__ inline v8bf cvt8(float4 a, float4 b) {
  v8bf p;
  p[0]=(__bf16)a.x; p[1]=(__bf16)a.y; p[2]=(__bf16)a.z; p[3]=(__bf16)a.w;
  p[4]=(__bf16)b.x; p[5]=(__bf16)b.y; p[6]=(__bf16)b.z; p[7]=(__bf16)b.w;
  return p;
}

// ---- Prep kernels ----------------------------------------------------------
// Combined proj+conv weights, stored TRANSPOSED: mkT[oc][tap*256+d]
__global__ void k_prep_mk(const float* __restrict__ Wproj,
                          const float* __restrict__ convw,
                          __bf16* __restrict__ mkT) {
  int idx = blockIdx.x * THREADS + threadIdx.x;           // over 256*768
  if (idx >= DDIM * GD) return;
  int d = idx / GD, oc = idx - d * GD;
  int g = oc >> 8;
  float a0 = 0.f, a1 = 0.f, a2 = 0.f;
  for (int i = 0; i < DDIM; ++i) {
    float wp = Wproj[d * GD + (g << 8) + i];
    const float* cw = convw + ((size_t)oc * DDIM + i) * 3;
    a0 += wp * cw[0]; a1 += wp * cw[1]; a2 += wp * cw[2];
  }
  mkT[(size_t)oc * GD + 0 * DDIM + d] = (__bf16)a0;
  mkT[(size_t)oc * GD + 1 * DDIM + d] = (__bf16)a1;
  mkT[(size_t)oc * GD + 2 * DDIM + d] = (__bf16)a2;
}

// fp32 [K][N] -> bf16 transposed [N][K]
__global__ void k_f2bfT(const float* __restrict__ src, __bf16* __restrict__ dst,
                        int K, int N) {
  int idx = blockIdx.x * THREADS + threadIdx.x;
  if (idx >= K * N) return;
  int k = idx / N, n = idx - k * N;
  dst[(size_t)n * K + k] = (__bf16)src[idx];
}

__global__ void k_minpos(const int* __restrict__ pos, float* __restrict__ minp) {
  __shared__ int sd[THREADS];
  int b = blockIdx.x, t = threadIdx.x;
  int m = 0x7fffffff;
  for (int s = t; s < SDIM; s += THREADS) {
    int p = pos[b * SDIM + s];
    if (p != -1 && p < m) m = p;
  }
  sd[t] = m; __syncthreads();
  for (int o = THREADS / 2; o > 0; o >>= 1) {
    if (t < o) sd[t] = min(sd[t], sd[t + o]);
    __syncthreads();
  }
  if (t == 0) minp[b] = (sd[0] == 0x7fffffff) ? 0.f : (float)sd[0];
}

__global__ void k_rope(const float* __restrict__ emb, const int* __restrict__ pos,
                       const float* __restrict__ minp, __bf16* __restrict__ tbf) {
  int row = blockIdx.x;             // b*S+s
  int j   = threadIdx.x;            // 0..127 (pair index)
  int b   = row >> 12;
  int p   = pos[row];
  float adj   = (p == -1) ? -1.0f : ((float)p - minp[b]);
  float theta = __powf(10000.0f, -((float)j) * (1.0f / 128.0f));
  float sn, cs;
  sincosf(adj * theta, &sn, &cs);   // large-angle accurate
  size_t o = (size_t)row * DDIM + 2 * j;
  float x1 = emb[o], x2 = emb[o + 1];
  tbf[o]     = (__bf16)(x1 * cs - x2 * sn);
  tbf[o + 1] = (__bf16)(x1 * sn + x2 * cs);
}

// ---- GEMM kernels ----------------------------------------------------------
__global__ __launch_bounds__(THREADS)
void k_gemm_projconv(const float* __restrict__ emb, const int* __restrict__ pos,
                     const __bf16* __restrict__ mkT, const float* __restrict__ convb,
                     float* __restrict__ zout) {
  const int row0 = blockIdx.x * BM;      // 128 | 4096 -> tile stays in one b
  const int b = row0 >> 12, s0 = row0 & (SDIM - 1);
  const int tn = blockIdx.y;
  auto stageA = [&](__bf16* As, int k0) {        // fp32 source: load+convert
    int tap = k0 >> 8, dbase = k0 & 255;         // tap uniform per k-tile
#pragma unroll
    for (int i = 0; i < 2; ++i) {
      int cidx = (int)threadIdx.x + i * THREADS; // 0..511: 128 rows x 4 chunks
      int r = cidx >> 2, q = cidx & 3;
      int ss = s0 + r + tap - 1;
      __bf16* dst = As + r * LDA + q * 8;
      if (ss >= 0 && ss < SDIM) {
        const float4* src =
            (const float4*)(emb + ((size_t)(b * SDIM + ss)) * DDIM + dbase + q * 8);
        *(v8bf*)dst = cvt8(src[0], src[1]);
      } else {
        v8bf z; for (int e = 0; e < 8; ++e) z[e] = (__bf16)0.f;
        *(v8bf*)dst = z;
      }
    }
  };
  auto stageB = [&](__bf16* Bs, int k0) { stage_wt(Bs, mkT, GD, GD, tn * BN, k0); };
  auto epi = [&](int ml, int nl, float v) {
    int row = row0 + ml, col = tn * BN + nl;
    float m = (pos[row] != -1) ? 1.f : 0.f;
    zout[(size_t)row * GD + col] = (v + convb[col]) * m;
  };
  gemm_core(GD, stageA, stageB, epi);    // K = 3 taps * 256
}

__global__ __launch_bounds__(THREADS)
void k_gemm_ff1(const __bf16* __restrict__ tbf, const __bf16* __restrict__ w1T,
                const float* __restrict__ b1, float* __restrict__ h1) {
  const int row0 = blockIdx.x * BM, tn = blockIdx.y;
  auto stageA = [&](__bf16* As, int k0) { stage_act(As, tbf, DDIM, row0, k0); };
  auto stageB = [&](__bf16* Bs, int k0) { stage_wt(Bs, w1T, DDIM, DDIM, tn * BN, k0); };
  auto epi = [&](int ml, int nl, float v) {
    int row = row0 + ml, col = tn * BN + nl;
    h1[(size_t)row * DDIM + col] = v + b1[col];
  };
  gemm_core(DDIM, stageA, stageB, epi);
}

__global__ void k_ln_gelu(const float* __restrict__ h1, const float* __restrict__ lng,
                          const float* __restrict__ lnb, __bf16* __restrict__ gout) {
  __shared__ float sd[DDIM];
  int row = blockIdx.x, t = threadIdx.x;
  float v = h1[(size_t)row * DDIM + t];
  sd[t] = v; __syncthreads();
  for (int o = 128; o > 0; o >>= 1) { if (t < o) sd[t] += sd[t + o]; __syncthreads(); }
  float mu = sd[0] * (1.f / 256.f);
  __syncthreads();
  float dv = v - mu;
  sd[t] = dv * dv; __syncthreads();
  for (int o = 128; o > 0; o >>= 1) { if (t < o) sd[t] += sd[t + o]; __syncthreads(); }
  float var = sd[0] * (1.f / 256.f);
  float x = dv * rsqrtf(var + 1e-5f) * lng[t] + lnb[t];
  float g = 0.5f * x * (1.f + erff(x * 0.70710678118f));   // exact GELU
  gout[(size_t)row * DDIM + t] = (__bf16)g;
}

__global__ __launch_bounds__(THREADS)
void k_gemm_ff2(const __bf16* __restrict__ gbuf, const __bf16* __restrict__ w2T,
                const float* __restrict__ b2, const float* __restrict__ ffs,
                float* __restrict__ h2) {
  const int row0 = blockIdx.x * BM, tn = blockIdx.y;
  auto stageA = [&](__bf16* As, int k0) { stage_act(As, gbuf, DDIM, row0, k0); };
  auto stageB = [&](__bf16* Bs, int k0) { stage_wt(Bs, w2T, DDIM, ND2, tn * BN, k0); };
  auto epi = [&](int ml, int nl, float v) {
    int row = row0 + ml, col = tn * BN + nl;
    h2[(size_t)row * ND2 + col] = (v + b2[col]) * ffs[col];
  };
  gemm_core(DDIM, stageA, stageB, epi);
}

__global__ void k_hhat_norm(float* __restrict__ h2) {
  __shared__ float sd[ND2];
  int row = blockIdx.x, t = threadIdx.x;       // 512 threads, 2 orders x 256
  size_t base = (size_t)row * ND2;
  float v = h2[base + t];
  sd[t] = fabsf(v); __syncthreads();
  for (int o = 128; o > 0; o >>= 1) {
    if ((t & 255) < o) sd[t] += sd[t + o];
    __syncthreads();
  }
  float denom = sd[t & 256] + 1e-8f;
  h2[base + t] = v / denom;
}

// ---- FFT long conv ---------------------------------------------------------
__device__ inline void fft8192(float2* c, const float2* tw, int t, bool inv) {
  const int n = 8192;
  for (int lg = 13; lg >= 1; --lg) {
    int half = 1 << (lg - 1);
    int jsh  = 13 - lg;
    for (int idx = t; idx < (n >> 1); idx += THREADS) {
      int p  = idx & (half - 1);
      int i0 = ((idx >> (lg - 1)) << lg) + p;
      int i1 = i0 + half;
      float2 a = c[i0], b = c[i1];
      float2 su = make_float2(a.x + b.x, a.y + b.y);
      float2 di = make_float2(a.x - b.x, a.y - b.y);
      float2 w  = tw[p << jsh];
      float wi  = inv ? -w.y : w.y;
      c[i0] = su;
      c[i1] = make_float2(di.x * w.x - di.y * wi, di.x * wi + di.y * w.x);
    }
    __syncthreads();
  }
}

template <int ORDER>
__global__ void k_fftconv(const float* __restrict__ zconv, const float* __restrict__ h2n,
                          float* __restrict__ vbuf) {
  extern __shared__ float sraw[];
  float2* c  = (float2*)sraw;                  // 8192 complex = 64 KB
  float2* tw = c + 8192;                       // 4096 twiddles = 32 KB
  const int n = 8192, L = SDIM;
  int ch = blockIdx.x, b = ch >> 8, d = ch & 255, t = threadIdx.x;

  for (int j = t; j < (n >> 1); j += THREADS) {  // W_n^j, forward sign
    float sn, cs;
    __sincosf(-6.283185307179586f * (float)j * (1.0f / 8192.0f), &sn, &cs);
    tw[j] = make_float2(cs, sn);
  }
  // pack c = v + i*h  (one FFT computes both real spectra)
  for (int s = t; s < L; s += THREADS) {
    float xv = (ORDER == 0)
        ? zconv[((size_t)(b * SDIM + s)) * GD + 2 * DDIM + d]   // v0 = zs[G-1]
        : vbuf[((size_t)(b * DDIM + d)) * SDIM + s];
    float fv = h2n[((size_t)(b * SDIM + s)) * ND2 + ORDER * DDIM + d];
    c[s] = make_float2(xv, fv);
  }
  for (int s = L + t; s < n; s += THREADS) c[s] = make_float2(0.f, 0.f);
  __syncthreads();

  fft8192(c, tw, t, false);                    // DIF forward: natural -> bitrev

  for (int i = t; i < n; i += THREADS) {       // permute to natural order
    int j = __brev(i) >> 19;
    if (j > i) { float2 tmp = c[i]; c[i] = c[j]; c[j] = tmp; }
  }
  __syncthreads();

  // Hermitian split of packed spectra, multiply, repack Hermitian product
  for (int k = t; k <= (n >> 1); k += THREADS) {
    int nk = (n - k) & (n - 1);
    float2 Ck = c[k], Cn = c[nk];
    float Xr = 0.5f * (Ck.x + Cn.x), Xi = 0.5f * (Ck.y - Cn.y);
    float Fr = 0.5f * (Ck.y + Cn.y), Fi = 0.5f * (Cn.x - Ck.x);
    float Pr = Xr * Fr - Xi * Fi;
    float Pi = Xr * Fi + Xi * Fr;
    c[k] = make_float2(Pr, Pi);
    if (nk != k) c[nk] = make_float2(Pr, -Pi);
  }
  __syncthreads();

  fft8192(c, tw, t, true);                     // inverse (unnorm): natural -> bitrev

  // y[s] lives at bitrev(s); scale 1/n (ifft) * 1/n (reference's extra /n)
  const float s2 = 1.0f / ((float)n * (float)n);
  for (int s = t; s < L; s += THREADS) {
    float y    = c[__brev(s) >> 19].x * s2;
    float gate = zconv[((size_t)(b * SDIM + s)) * GD + ORDER * DDIM + d];  // masked already
    vbuf[((size_t)(b * DDIM + d)) * SDIM + s] = gate * y;
  }
}

// ---- Output GEMM (A = v^T, contiguous loads along s from (B,D,S)) ----------
__global__ __launch_bounds__(THREADS)
void k_gemm_out(const float* __restrict__ vbuf, const __bf16* __restrict__ woutT,
                const float* __restrict__ bout, const int* __restrict__ pos,
                float* __restrict__ out) {
  const int row0 = blockIdx.x * BM;
  const int b = row0 >> 12, s0 = row0 & (SDIM - 1);
  const int tn = blockIdx.y;
  auto stageA = [&](__bf16* As, int k0) {        // fp32 transposed source
    int kk = (int)threadIdx.x & 31, rc = (int)threadIdx.x >> 5;  // 8 row-chunks of 16
    const float4* src = (const float4*)(vbuf +
        ((size_t)(b * DDIM + k0 + kk)) * SDIM + s0 + rc * 16);
    float4 x0 = src[0], x1 = src[1], x2 = src[2], x3 = src[3];
    float vals[16] = {x0.x,x0.y,x0.z,x0.w, x1.x,x1.y,x1.z,x1.w,
                      x2.x,x2.y,x2.z,x2.w, x3.x,x3.y,x3.z,x3.w};
#pragma unroll
    for (int j = 0; j < 16; ++j) As[(rc * 16 + j) * LDA + kk] = (__bf16)vals[j];
  };
  auto stageB = [&](__bf16* Bs, int k0) { stage_wt(Bs, woutT, DDIM, DDIM, tn * BN, k0); };
  auto epi = [&](int ml, int nl, float v) {
    int row = row0 + ml, col = tn * BN + nl;
    float m = (pos[row] != -1) ? 1.f : 0.f;
    out[(size_t)row * DDIM + col] = (v + bout[col]) * m;
  };
  gemm_core(DDIM, stageA, stageB, epi);
}

// ---- Host ------------------------------------------------------------------
extern "C" void kernel_launch(void* const* d_in, const int* in_sizes, int n_in,
                              void* d_out, int out_size, void* d_ws, size_t ws_size,
                              hipStream_t stream) {
  const float* emb   = (const float*)d_in[0];
  const int*   pos   = (const int*)d_in[1];
  const float* Wproj = (const float*)d_in[2];
  const float* convw = (const float*)d_in[3];
  const float* convb = (const float*)d_in[4];
  const float* W1    = (const float*)d_in[5];
  const float* b1    = (const float*)d_in[6];
  const float* lng   = (const float*)d_in[7];
  const float* lnb   = (const float*)d_in[8];
  const float* W2    = (const float*)d_in[9];
  const float* b2    = (const float*)d_in[10];
  const float* ffs   = (const float*)d_in[11];
  const float* Wout  = (const float*)d_in[12];
  const float* bout  = (const float*)d_in[13];
  float* out = (float*)d_out;

  char* w = (char*)d_ws;
  size_t off = 0;
  auto take = [&](size_t bytes) -> void* {
    void* p = (void*)(w + off);
    off += (bytes + 255) & ~(size_t)255;
    return p;
  };
  __bf16* mkT    = (__bf16*)take((size_t)GD * GD * 2);           // 1.18 MB [oc][768]
  __bf16* w1T    = (__bf16*)take((size_t)DDIM * DDIM * 2);       // 128 KB  [n][k]
  __bf16* w2T    = (__bf16*)take((size_t)ND2 * DDIM * 2);        // 256 KB  [n][k]
  __bf16* woutT  = (__bf16*)take((size_t)DDIM * DDIM * 2);       // 128 KB  [n][k]
  float*  minp   = (float*)take((size_t)BDIM * 4);
  __bf16* tbf    = (__bf16*)take((size_t)BDIM * SDIM * DDIM * 2);   // 16.8 MB (aliased by gbuf)
  float*  zconv  = (float*)take((size_t)BDIM * SDIM * GD * 4);      // 100.7 MB
  float*  h1     = (float*)take((size_t)BDIM * SDIM * DDIM * 4);    // 33.6 MB (aliased by vbuf)
  float*  h2     = (float*)take((size_t)BDIM * SDIM * ND2 * 4);     // 67.1 MB
  __bf16* gbuf = tbf;   // t consumed by ff1 before ln_gelu writes gbuf
  float*  vbuf = h1;    // h1 consumed by ln_gelu before fftconv writes vbuf
  (void)ws_size; (void)n_in; (void)in_sizes; (void)out_size;

  const int BS = BDIM * SDIM;  // 32768

  k_prep_mk<<<(DDIM * GD + THREADS - 1) / THREADS, THREADS, 0, stream>>>(Wproj, convw, mkT);
  k_f2bfT<<<(DDIM * DDIM + THREADS - 1) / THREADS, THREADS, 0, stream>>>(W1, w1T, DDIM, DDIM);
  k_f2bfT<<<(DDIM * ND2 + THREADS - 1) / THREADS, THREADS, 0, stream>>>(W2, w2T, DDIM, ND2);
  k_f2bfT<<<(DDIM * DDIM + THREADS - 1) / THREADS, THREADS, 0, stream>>>(Wout, woutT, DDIM, DDIM);
  k_minpos<<<BDIM, THREADS, 0, stream>>>(pos, minp);
  k_rope<<<BS, 128, 0, stream>>>(emb, pos, minp, tbf);

  k_gemm_projconv<<<dim3(BS / BM, GD / BN), THREADS, 0, stream>>>(emb, pos, mkT, convb, zconv);
  k_gemm_ff1<<<dim3(BS / BM, DDIM / BN), THREADS, 0, stream>>>(tbf, w1T, b1, h1);
  k_ln_gelu<<<BS, DDIM, 0, stream>>>(h1, lng, lnb, gbuf);
  k_gemm_ff2<<<dim3(BS / BM, ND2 / BN), THREADS, 0, stream>>>(gbuf, w2T, b2, ffs, h2);
  k_hhat_norm<<<BS, ND2, 0, stream>>>(h2);

  const size_t fft_lds = (8192 + 4096) * sizeof(float2);   // 96 KB
  k_fftconv<0><<<BDIM * DDIM, THREADS, fft_lds, stream>>>(zconv, h2, vbuf);
  k_fftconv<1><<<BDIM * DDIM, THREADS, fft_lds, stream>>>(zconv, h2, vbuf);

  k_gemm_out<<<dim3(BS / BM, DDIM / BN), THREADS, 0, stream>>>(vbuf, woutT, bout, pos, out);
}
